// MS_SSA_Conv_18305150615568
// MI455X (gfx1250) — matrix operand order repaired and checked by reading, not compiled, not run on hardware
//
#include <hip/hip_runtime.h>
#include <hip/hip_bf16.h>

// MS-SSA spiking self-attention block, fused for gfx1250 (MI455X).
// Pipeline:
//   K0: weight pre-convert f32 -> f16 (done once; GEMM A-operands load f16)
//   K1: shortcut LIF  x(f32) -> xs spikes (f16)                [elementwise, T in regs]
//   K2: q/k/v = LIF(BN(W @ xs))  -- one wave per 16x16 tile, WMMA f16,
//       all 4 time-steps accumulated simultaneously (12 accumulators),
//       BN+LIF fused in epilogue, spikes written as f16.
//   K3: kv = LIF(sum_N k*v, 0.5) -- async global->LDS staging (ASYNCcnt) +
//       LDS tree reduction, T recurrence at lane 0.
//   K4: out = BN(Wp @ (q*kv) + b) + x -- att fused into B-operand load.

typedef __attribute__((ext_vector_type(16))) _Float16 v16h;
typedef __attribute__((ext_vector_type(8)))  _Float16 v8h;
typedef __attribute__((ext_vector_type(8)))  float    v8f;

#define T_ 4
#define B_ 16
#define C_ 512
#define N_ 256
static constexpr size_t BCN  = (size_t)B_ * C_ * N_;   // 2,097,152
static constexpr size_t TBCN = (size_t)T_ * BCN;       // 8,388,608
static constexpr size_t WELEM = (size_t)C_ * C_;       // 262,144 per matrix

static __device__ inline v8f wmma_f16(v16h a, v16h b, v8f c) {
  // D = A(16x32 f16) * B(32x16 f16) + C(16x16 f32)
  return __builtin_amdgcn_wmma_f32_16x16x32_f16(
      /*neg_a=*/false, a, /*neg_b=*/false, b,
      /*c_mod=*/(short)0, c, /*reuse_a=*/false, /*reuse_b=*/false);
}

// Load a 16-element f16 A-operand slice for this lane from an f16 row-major
// weight matrix. ISA layout (16-bit A 16x32): lane (mod 16) = row M; lanes<16
// hold K {0..7,16..23}, lanes>=16 hold K {8..15,24..31}. Caller passes
// p = &W[row*C + k0 + hi*8]; elements 0..7 = p[0..7], 8..15 = p[16..23].
static __device__ inline v16h load_a_h(const _Float16* __restrict__ p) {
  v8h lo = *(const v8h*)p;          // 16B aligned: offset multiple of 8 halves
  v8h hi = *(const v8h*)(p + 16);
  v16h a;
#pragma unroll
  for (int j = 0; j < 8; ++j) { a[j] = lo[j]; a[8 + j] = hi[j]; }
  return a;
}

// ---------------- K0: weight f32 -> f16 (all four matrices) ----------------
__global__ __launch_bounds__(256) void cvt_w_kernel(
    const float* __restrict__ qw, const float* __restrict__ kw,
    const float* __restrict__ vw, const float* __restrict__ pw,
    _Float16* __restrict__ qh, _Float16* __restrict__ kh,
    _Float16* __restrict__ vh, _Float16* __restrict__ ph) {
  size_t i = (size_t)blockIdx.x * blockDim.x + threadIdx.x;
  if (i >= WELEM) return;
  qh[i] = (_Float16)qw[i];
  kh[i] = (_Float16)kw[i];
  vh[i] = (_Float16)vw[i];
  ph[i] = (_Float16)pw[i];
}

// ---------------- K1: shortcut LIF ----------------
__global__ __launch_bounds__(256) void lif_in_kernel(
    const float* __restrict__ x, _Float16* __restrict__ xs) {
  size_t i = (size_t)blockIdx.x * blockDim.x + threadIdx.x;
  if (i >= BCN) return;
  float v = 0.0f;
#pragma unroll
  for (int t = 0; t < T_; ++t) {
    float xv = x[t * BCN + i];
    v = 0.5f * (v + xv);                       // v += (x - v)/TAU, TAU=2
    float s = (v >= 1.0f) ? 1.0f : 0.0f;       // spike, v_th = 1
    xs[t * BCN + i] = (_Float16)s;
    v = (s > 0.0f) ? 0.0f : v;                 // hard reset
  }
}

// ---------------- K2: q/k/v GEMM + BN + LIF ----------------
__global__ __launch_bounds__(32) void qkv_kernel(
    const _Float16* __restrict__ xs,
    const _Float16* __restrict__ qw, const float* __restrict__ qg,
    const float* __restrict__ qbe, const float* __restrict__ qme,
    const float* __restrict__ qva,
    const _Float16* __restrict__ kw, const float* __restrict__ kg,
    const float* __restrict__ kbe, const float* __restrict__ kme,
    const float* __restrict__ kva,
    const _Float16* __restrict__ vw, const float* __restrict__ vg,
    const float* __restrict__ vbe, const float* __restrict__ vme,
    const float* __restrict__ vva,
    _Float16* __restrict__ qs, _Float16* __restrict__ ks,
    _Float16* __restrict__ vs) {
  const int lane = threadIdx.x;
  const int col  = lane & 15;
  const int hi   = lane >> 4;
  const int n0   = blockIdx.x << 4;
  const int o0   = blockIdx.y << 4;
  const int b    = blockIdx.z;
  const int arow = o0 + col;                    // A-operand row for this lane

  const _Float16* W[3] = {qw, kw, vw};
  const float* G[3]    = {qg, kg, vg};
  const float* BE[3]   = {qbe, kbe, vbe};
  const float* ME[3]   = {qme, kme, vme};
  const float* VA[3]   = {qva, kva, vva};
  _Float16* OUT[3]     = {qs, ks, vs};

  // BN fold: y = acc*inv + add, per C/D row (row = o0 + hi*8 + r)
  float bninv[3][8], bnadd[3][8];
#pragma unroll
  for (int br = 0; br < 3; ++br) {
#pragma unroll
    for (int r = 0; r < 8; ++r) {
      int row = o0 + hi * 8 + r;
      float iv = G[br][row] * rsqrtf(VA[br][row] + 1e-5f);
      bninv[br][r] = iv;
      bnadd[br][r] = BE[br][row] - ME[br][row] * iv;
    }
  }

  v8f acc[12];                                  // [branch][t], all t live
  v8f zero = {};
#pragma unroll
  for (int i = 0; i < 12; ++i) acc[i] = zero;

  const size_t abase = (size_t)arow * C_ + hi * 8;

  for (int k0 = 0; k0 < C_; k0 += 32) {
    // A tiles: one f16 load per branch per k-step, reused by 4 time steps
    v16h a0 = load_a_h(W[0] + abase + k0);
    v16h a1 = load_a_h(W[1] + abase + k0);
    v16h a2 = load_a_h(W[2] + abase + k0);
    // Prefetch next k-step's weight rows (global_prefetch_b8)
    __builtin_prefetch(W[0] + abase + k0 + 32, 0, 1);
    __builtin_prefetch(W[1] + abase + k0 + 32, 0, 1);
    __builtin_prefetch(W[2] + abase + k0 + 32, 0, 1);
#pragma unroll
    for (int t = 0; t < T_; ++t) {
      // B-operand: lane holds column N = n0+col, K in table order
      v16h bt;
#pragma unroll
      for (int s = 0; s < 16; ++s) {
        int kk = k0 + hi * 8 + (s & 7) + ((s >> 3) << 4);
        bt[s] = xs[((size_t)(t * B_ + b) * C_ + kk) * N_ + n0 + col];
      }
      acc[0 * 4 + t] = wmma_f16(a0, bt, acc[0 * 4 + t]);
      acc[1 * 4 + t] = wmma_f16(a1, bt, acc[1 * 4 + t]);
      acc[2 * 4 + t] = wmma_f16(a2, bt, acc[2 * 4 + t]);
    }
  }

  // Epilogue: BN + LIF recurrence over t (state in registers), emit spikes
  float vst[3][8];
#pragma unroll
  for (int br = 0; br < 3; ++br)
#pragma unroll
    for (int r = 0; r < 8; ++r) vst[br][r] = 0.0f;

#pragma unroll
  for (int t = 0; t < T_; ++t) {
#pragma unroll
    for (int br = 0; br < 3; ++br) {
#pragma unroll
      for (int r = 0; r < 8; ++r) {
        float u  = acc[br * 4 + t][r] * bninv[br][r] + bnadd[br][r];
        float nv = 0.5f * (vst[br][r] + u);
        float s  = (nv >= 1.0f) ? 1.0f : 0.0f;
        int row  = o0 + hi * 8 + r;
        OUT[br][((size_t)(t * B_ + b) * C_ + row) * N_ + n0 + col] =
            (_Float16)s;
        vst[br][r] = (s > 0.0f) ? 0.0f : nv;
      }
    }
  }
}

// ---------------- K3: kv = LIF(sum_N k*v, 0.5) ----------------
// k/v spike rows are staged into LDS with the CDNA5 async global->LDS path
// (GLOBAL_LOAD_ASYNC_TO_LDS_B32, tracked by ASYNCcnt), then tree-reduced.
__global__ __launch_bounds__(256) void kv_kernel(
    const _Float16* __restrict__ ks, const _Float16* __restrict__ vs,
    _Float16* __restrict__ kvh, float* __restrict__ kv_out) {
  __shared__ _Float16 kbuf[N_];
  __shared__ _Float16 vbuf[N_];
  __shared__ float red[256];
  const int n  = threadIdx.x;
  const int bc = blockIdx.x;
  const int b  = bc >> 9;          // bc / 512
  const int c  = bc & 511;
  float vstate = 0.0f;
  for (int t = 0; t < T_; ++t) {
    size_t rowbase = ((size_t)(t * B_ + b) * C_ + c) * N_;
    // Async stage: threads 0..127 copy k row (4B each), 128..255 copy v row.
    {
      const _Float16* gp;
      unsigned ldsoff;
      if (n < 128) {
        gp = ks + rowbase + (size_t)n * 2;
        ldsoff = (unsigned)(uintptr_t)(&kbuf[n * 2]);   // low 32b = LDS offset
      } else {
        gp = vs + rowbase + (size_t)(n - 128) * 2;
        ldsoff = (unsigned)(uintptr_t)(&vbuf[(n - 128) * 2]);
      }
      asm volatile("global_load_async_to_lds_b32 %0, %1, off"
                   :: "v"(ldsoff), "v"(gp) : "memory");
      asm volatile("s_wait_asynccnt 0x0" ::: "memory");
    }
    __syncthreads();
    red[n] = (float)kbuf[n] * (float)vbuf[n];
    __syncthreads();
    for (int sft = 128; sft > 0; sft >>= 1) {
      if (n < sft) red[n] += red[n + sft];
      __syncthreads();
    }
    if (n == 0) {
      float u = red[0];
      vstate  = 0.5f * (vstate + u);
      float s = (vstate >= 0.5f) ? 1.0f : 0.0f;   // v_th = 0.5
      size_t o = (size_t)(t * B_ + b) * C_ + c;
      kvh[o]    = (_Float16)s;
      kv_out[o] = s;                               // second reference output
      vstate = (s > 0.0f) ? 0.0f : vstate;
    }
    __syncthreads();
  }
}

// ---------------- K4: proj GEMM + bias + BN + residual ----------------
__global__ __launch_bounds__(32) void proj_kernel(
    const _Float16* __restrict__ qs, const _Float16* __restrict__ kvh,
    const _Float16* __restrict__ pw, const float* __restrict__ pb,
    const float* __restrict__ pg, const float* __restrict__ pbe,
    const float* __restrict__ pme, const float* __restrict__ pva,
    const float* __restrict__ x, float* __restrict__ out) {
  const int lane = threadIdx.x;
  const int col  = lane & 15;
  const int hi   = lane >> 4;
  const int n0   = blockIdx.x << 4;
  const int o0   = blockIdx.y << 4;
  const int b    = blockIdx.z & 15;
  const int t    = blockIdx.z >> 4;
  const int arow = o0 + col;

  // Fold bias into BN shift: BN(z + pb) = z*inv + (pb*inv + beta - mean*inv)
  float bninv[8], bnadd[8];
#pragma unroll
  for (int r = 0; r < 8; ++r) {
    int row = o0 + hi * 8 + r;
    float iv = pg[row] * rsqrtf(pva[row] + 1e-5f);
    bninv[r] = iv;
    bnadd[r] = pb[row] * iv + pbe[row] - pme[row] * iv;
  }

  const size_t abase = (size_t)arow * C_ + hi * 8;

  v8f acc = {};
  for (int k0 = 0; k0 < C_; k0 += 32) {
    v16h a = load_a_h(pw + abase + k0);
    __builtin_prefetch(pw + abase + k0 + 32, 0, 1);
    // att = q * kv fused into B-operand: element j (K in table order) scaled by kv[K]
    v16h bt;
#pragma unroll
    for (int j = 0; j < 16; ++j) {
      int kk = k0 + hi * 8 + (j & 7) + ((j >> 3) << 4);
      _Float16 sc = kvh[(size_t)(t * B_ + b) * C_ + kk];
      bt[j] = qs[((size_t)(t * B_ + b) * C_ + kk) * N_ + n0 + col] * sc;
    }
    acc = wmma_f16(a, bt, acc);
  }

#pragma unroll
  for (int r = 0; r < 8; ++r) {
    int row = o0 + hi * 8 + r;
    size_t idx = ((size_t)(t * B_ + b) * C_ + row) * N_ + n0 + col;
    out[idx] = acc[r] * bninv[r] + bnadd[r] + x[idx];   // + identity
  }
}

extern "C" void kernel_launch(void* const* d_in, const int* in_sizes, int n_in,
                              void* d_out, int out_size, void* d_ws,
                              size_t ws_size, hipStream_t stream) {
  (void)in_sizes; (void)n_in; (void)out_size; (void)ws_size;
  const float* x   = (const float*)d_in[0];
  const float* qw  = (const float*)d_in[1];
  const float* qg  = (const float*)d_in[2];
  const float* qbe = (const float*)d_in[3];
  const float* qme = (const float*)d_in[4];
  const float* qva = (const float*)d_in[5];
  const float* kw  = (const float*)d_in[6];
  const float* kg  = (const float*)d_in[7];
  const float* kbe = (const float*)d_in[8];
  const float* kme = (const float*)d_in[9];
  const float* kva = (const float*)d_in[10];
  const float* vw  = (const float*)d_in[11];
  const float* vg  = (const float*)d_in[12];
  const float* vbe = (const float*)d_in[13];
  const float* vme = (const float*)d_in[14];
  const float* vva = (const float*)d_in[15];
  const float* pw  = (const float*)d_in[16];
  const float* pb  = (const float*)d_in[17];
  const float* pg  = (const float*)d_in[18];
  const float* pbe = (const float*)d_in[19];
  const float* pme = (const float*)d_in[20];
  const float* pva = (const float*)d_in[21];

  // Workspace: 4 spike tensors (f16) + kv spikes (f16) + 4 f16 weights.
  char* wsb = (char*)d_ws;
  const size_t SPIKE_BYTES = TBCN * sizeof(_Float16);          // 16 MB each
  const size_t KV_BYTES    = (size_t)T_ * B_ * C_ * sizeof(_Float16);
  const size_t W_BYTES     = WELEM * sizeof(_Float16);         // 0.5 MB each
  _Float16* xs  = (_Float16*)(wsb + 0 * SPIKE_BYTES);
  _Float16* qsp = (_Float16*)(wsb + 1 * SPIKE_BYTES);
  _Float16* ksp = (_Float16*)(wsb + 2 * SPIKE_BYTES);
  _Float16* vsp = (_Float16*)(wsb + 3 * SPIKE_BYTES);
  _Float16* kvh = (_Float16*)(wsb + 4 * SPIKE_BYTES);
  char* wh      = wsb + 4 * SPIKE_BYTES + KV_BYTES;
  _Float16* qwh = (_Float16*)(wh + 0 * W_BYTES);
  _Float16* kwh = (_Float16*)(wh + 1 * W_BYTES);
  _Float16* vwh = (_Float16*)(wh + 2 * W_BYTES);
  _Float16* pwh = (_Float16*)(wh + 3 * W_BYTES);

  float* out    = (float*)d_out;
  float* kv_out = out + TBCN;   // second tuple element [T,B,h,1,d] = [T,B,512]

  cvt_w_kernel<<<(unsigned)((WELEM + 255) / 256), 256, 0, stream>>>(
      qw, kw, vw, pw, qwh, kwh, vwh, pwh);

  lif_in_kernel<<<(unsigned)((BCN + 255) / 256), 256, 0, stream>>>(x, xs);

  qkv_kernel<<<dim3(N_ / 16, C_ / 16, B_), 32, 0, stream>>>(
      xs, qwh, qg, qbe, qme, qva, kwh, kg, kbe, kme, kva, vwh, vg, vbe, vme,
      vva, qsp, ksp, vsp);

  kv_kernel<<<B_ * C_, 256, 0, stream>>>(ksp, vsp, kvh, kv_out);

  proj_kernel<<<dim3(N_ / 16, C_ / 16, B_ * T_), 32, 0, stream>>>(
      qsp, kvh, pwh, pb, pg, pbe, pme, pva, x, out);
}
